// hyperedge_atten_77077483094552
// MI455X (gfx1250) — matrix operand order repaired and checked by reading, not compiled, not run on hardware
//
#include <hip/hip_runtime.h>
#include <hip/hip_bf16.h>

// ---------------- problem constants (fixed by the reference harness) --------
#define N_NODES 20000
#define N_EDGES 1000
#define KMEM    32
#define FDIM    512
#define NHID    2000
#define NTILES  (NHID / 16)      // 125 exact
#define ECHUNK  125              // edges per chunk -> sf/res chunk = 64 MB, L2 resident
#define NCHUNK  (N_EDGES / ECHUNK)
#define HT      80               // h-tile for the attention combine (2000 % 80 == 0)
#define SLOPE   0.2f

typedef __bf16 bf16_t;
typedef bf16_t v16bf __attribute__((ext_vector_type(16)));
typedef float  v8f   __attribute__((ext_vector_type(8)));

// ---------------- fragment helpers ------------------------------------------
__device__ __forceinline__ v16bf load_frag2(const bf16_t* plo, const bf16_t* phi) {
    v16bf v;
    reinterpret_cast<uint4*>(&v)[0] = *reinterpret_cast<const uint4*>(plo);
    reinterpret_cast<uint4*>(&v)[1] = *reinterpret_cast<const uint4*>(phi);
    return v;
}

__device__ __forceinline__ v8f wmma_bf16(v16bf a, v16bf b, v8f c) {
    // (neg_a, A, neg_b, B, c_mod, C, reuse_a, reuse_b)
    return __builtin_amdgcn_wmma_f32_16x16x32_bf16(false, a, false, b, (short)0, c, false, false);
}

// ---------------- kernel 0: init y = breg, cnt = 0 --------------------------
__global__ void k_init(float* __restrict__ y, int* __restrict__ cnt,
                       const float* __restrict__ breg) {
    int i = blockIdx.x * blockDim.x + threadIdx.x;
    if (i < N_NODES) y[i] = breg[0];
    if (i < N_EDGES) cnt[i] = 0;
}

// ---------------- kernel 1: membership scan of H ----------------------------
// H is (N, E) row-major; one thread per element, coalesced along E.
// Unordered member list is fine: the per-edge math is permutation invariant.
__global__ void k_member(const float* __restrict__ H, int* __restrict__ cnt,
                         int* __restrict__ idx) {
    size_t i = (size_t)blockIdx.x * blockDim.x + threadIdx.x;
    if (i >= (size_t)N_NODES * N_EDGES) return;
    if (H[i] > 0.5f) {
        int n = (int)(i / N_EDGES);
        int e = (int)(i % N_EDGES);
        int p = atomicAdd(&cnt[e], 1);
        if (p < KMEM) idx[e * KMEM + p] = n;   // <= 32 ones per column by construction
    }
}

// ---------------- kernel 2: W1/Wres -> bf16, transposed to (NHID, F) --------
__global__ void k_convw(const float* __restrict__ W1, const float* __restrict__ Wres,
                        bf16_t* __restrict__ W1t, bf16_t* __restrict__ Wrt) {
    int i = blockIdx.x * blockDim.x + threadIdx.x;       // i = n*F + k
    if (i >= NHID * FDIM) return;
    int n = i / FDIM, k = i % FDIM;
    W1t[i] = (bf16_t)W1[(size_t)k * NHID + n];
    Wrt[i] = (bf16_t)Wres[(size_t)k * NHID + n];
}

// ---------------- kernel 3: gather x rows -> bf16 xg (zero padded) ----------
__global__ void k_gather(const float* __restrict__ x, const int* __restrict__ idx,
                         const int* __restrict__ cnt, bf16_t* __restrict__ xg) {
    int row  = blockIdx.x * 4 + (threadIdx.x >> 6);      // (e,k) flat row
    int sub  = threadIdx.x & 63;
    int e = row / KMEM, k = row % KMEM;
    bool valid = (k < cnt[e]);
    int node = valid ? idx[row] : 0;
    const float* src = x + (size_t)node * FDIM;
    bf16_t* dst = xg + (size_t)row * FDIM;
    int f0 = sub * 8;
    #pragma unroll
    for (int j = 0; j < 8; ++j)
        dst[f0 + j] = valid ? (bf16_t)src[f0 + j] : (bf16_t)0.0f;
}

// ---------------- kernel 4: WMMA GEMMs: sf = xg@W1+b1, res = xg@Wres+bres ---
// grid (16, ECHUNK), 256 threads = 8 waves; wave w owns ntile = bx*8+w.
// Each wave: 32x16 output tile for both sf and res; K=512 in 16 steps of 32.
__global__ void __launch_bounds__(256)
k_gemm(const bf16_t* __restrict__ xg, const bf16_t* __restrict__ W1t,
       const bf16_t* __restrict__ Wrt, const float* __restrict__ b1,
       const float* __restrict__ bres, float* __restrict__ sf,
       float* __restrict__ res, int e0) {
    int wave = threadIdx.x >> 5;
    int lane = threadIdx.x & 31;
    int ntile = blockIdx.x * 8 + wave;
    if (ntile >= NTILES) return;                          // wave-uniform: EXEC stays all-1
    int eloc = blockIdx.y;
    int e = e0 + eloc;
    int g = lane >> 4, r = lane & 15;
    int n0 = ntile * 16;

    const bf16_t* arow0 = xg + ((size_t)e * KMEM + r) * FDIM;        // M = r
    const bf16_t* arow1 = xg + ((size_t)e * KMEM + r + 16) * FDIM;   // M = r+16
    const bf16_t* brow1 = W1t + (size_t)(n0 + r) * FDIM;             // B[K][n] = W1t[n*F+K]
    const bf16_t* brow2 = Wrt + (size_t)(n0 + r) * FDIM;

    v8f s0 = {}, s1 = {}, r0 = {}, r1 = {};
    #pragma unroll 4
    for (int kk = 0; kk < FDIM; kk += 32) {
        // A frag (16-bit A 16x32 layout): elems 0..7 -> K = kk+8g.., 8..15 -> K = kk+16+8g..
        int klo = kk + 8 * g;
        int khi = kk + 16 + 8 * g;
        v16bf A0 = load_frag2(arow0 + klo, arow0 + khi);
        v16bf A1 = load_frag2(arow1 + klo, arow1 + khi);
        // B frag: same K order as A: elems 0..7 -> K = kk+8g.., 8..15 -> K = kk+16+8g..
        int kb = kk + 8 * g;
        v16bf B1 = load_frag2(brow1 + kb, brow1 + kb + 16);
        v16bf B2 = load_frag2(brow2 + kb, brow2 + kb + 16);
        s0 = wmma_bf16(A0, B1, s0);
        s1 = wmma_bf16(A1, B1, s1);
        r0 = wmma_bf16(A0, B2, r0);
        r1 = wmma_bf16(A1, B2, r1);
    }
    // C/D layout: VGPR v, lane-group g, col n = r  ->  row M = v + 8g (+16 for tile1)
    float bb1 = b1[n0 + r], bb2 = bres[n0 + r];
    float* sfe = sf + (size_t)eloc * KMEM * NHID;
    float* rse = res + (size_t)eloc * KMEM * NHID;
    #pragma unroll
    for (int v = 0; v < 8; ++v) {
        int m0 = v + 8 * g;
        sfe[(size_t)m0 * NHID + n0 + r]        = s0[v] + bb1;
        rse[(size_t)m0 * NHID + n0 + r]        = r0[v] + bb2;
        sfe[(size_t)(m0 + 16) * NHID + n0 + r] = s1[v] + bb1;
        rse[(size_t)(m0 + 16) * NHID + n0 + r] = r1[v] + bb2;
    }
}

// ---------------- kernel 5: attention + ELU + Wreg dot + scatter ------------
// one 256-thread block per edge; thread group of 8 owns row k = t>>3.
__global__ void __launch_bounds__(256)
k_attn(const float* __restrict__ sf, const float* __restrict__ res,
       const float* __restrict__ a1, const float* __restrict__ ba1,
       const float* __restrict__ a2, const float* __restrict__ ba2,
       const float* __restrict__ wreg, const int* __restrict__ idx,
       const int* __restrict__ cnt, float* __restrict__ y, int e0) {
    __shared__ float f1s[KMEM], f2s[KMEM];
    __shared__ float coefs[KMEM][KMEM + 1];
    __shared__ float tile[KMEM][HT];
    __shared__ float wr[HT];

    int eloc = blockIdx.x;
    int e = e0 + eloc;
    const float* sfe = sf + (size_t)eloc * KMEM * NHID;
    const float* rse = res + (size_t)eloc * KMEM * NHID;
    int t = threadIdx.x;
    int k = t >> 3, l8 = t & 7;
    int m = cnt[e];

    // --- f1/f2 = sf . a1/a2 + bias (8 lanes per row, shuffle reduce) ---
    float s1 = 0.f, s2 = 0.f;
    for (int h = l8; h < NHID; h += 8) {
        float s = sfe[(size_t)k * NHID + h];
        s1 += s * a1[h];
        s2 += s * a2[h];
    }
    s1 += __shfl_xor(s1, 1); s1 += __shfl_xor(s1, 2); s1 += __shfl_xor(s1, 4);
    s2 += __shfl_xor(s2, 1); s2 += __shfl_xor(s2, 2); s2 += __shfl_xor(s2, 4);
    if (l8 == 0) { f1s[k] = s1 + ba1[0]; f2s[k] = s2 + ba2[0]; }
    __syncthreads();

    // --- masked LeakyReLU softmax, row per thread (t < 32) ---
    if (t < KMEM) {
        float fi = f1s[t];
        float lg[KMEM];
        float mx = -1e30f;
        #pragma unroll
        for (int j = 0; j < KMEM; ++j) {
            float L = fi + f2s[j];
            L = (L > 0.f) ? L : SLOPE * L;
            if (j >= m) L = -1e9f;
            lg[j] = L;
            mx = fmaxf(mx, L);
        }
        float sum = 0.f;
        #pragma unroll
        for (int j = 0; j < KMEM; ++j) { float ex = __expf(lg[j] - mx); lg[j] = ex; sum += ex; }
        float inv = 1.f / sum;
        #pragma unroll
        for (int j = 0; j < KMEM; ++j) coefs[t][j] = lg[j] * inv;
    }
    __syncthreads();

    // --- out = elu(coefs@sf + res); ck = out . wreg  (tiled through LDS) ---
    float ck = 0.f;
    for (int hb = 0; hb < NHID; hb += HT) {
        for (int i = t; i < KMEM * HT; i += 256) {
            int q = i / HT, hh = i % HT;
            tile[q][hh] = sfe[(size_t)q * NHID + hb + hh];
        }
        if (t < HT) wr[t] = wreg[hb + t];
        __syncthreads();
        for (int hh = l8; hh < HT; hh += 8) {
            float rsum = 0.f;
            #pragma unroll
            for (int q = 0; q < KMEM; ++q) rsum += coefs[k][q] * tile[q][hh];
            float o = rsum + rse[(size_t)k * NHID + hb + hh];
            o = (o > 0.f) ? o : (__expf(o) - 1.f);   // ELU (alpha=1)
            ck += o * wr[hh];
        }
        __syncthreads();
    }
    ck += __shfl_xor(ck, 1); ck += __shfl_xor(ck, 2); ck += __shfl_xor(ck, 4);
    if (l8 == 0 && k < m) atomicAdd(&y[idx[e * KMEM + k]], ck);
}

// ---------------- host launcher ---------------------------------------------
extern "C" void kernel_launch(void* const* d_in, const int* in_sizes, int n_in,
                              void* d_out, int out_size, void* d_ws, size_t ws_size,
                              hipStream_t stream) {
    (void)in_sizes; (void)n_in; (void)out_size; (void)ws_size;
    const float* x    = (const float*)d_in[0];
    const float* H    = (const float*)d_in[1];
    const float* W1   = (const float*)d_in[2];
    const float* b1   = (const float*)d_in[3];
    const float* a1   = (const float*)d_in[4];
    const float* ba1  = (const float*)d_in[5];
    const float* a2   = (const float*)d_in[6];
    const float* ba2  = (const float*)d_in[7];
    const float* Wres = (const float*)d_in[8];
    const float* bres = (const float*)d_in[9];
    const float* Wreg = (const float*)d_in[10];
    const float* breg = (const float*)d_in[11];
    float* y = (float*)d_out;

    char* base = (char*)d_ws;
    size_t off = 0;
    auto take = [&](size_t bytes) -> void* {
        void* p = base + off;
        off += (bytes + 255) & ~(size_t)255;
        return p;
    };
    int*    cnt = (int*)take((size_t)N_EDGES * 4);
    int*    idx = (int*)take((size_t)N_EDGES * KMEM * 4);
    bf16_t* W1t = (bf16_t*)take((size_t)NHID * FDIM * 2);
    bf16_t* Wrt = (bf16_t*)take((size_t)NHID * FDIM * 2);
    bf16_t* xg  = (bf16_t*)take((size_t)N_EDGES * KMEM * FDIM * 2);
    float*  sf  = (float*)take((size_t)ECHUNK * KMEM * NHID * 4);
    float*  res = (float*)take((size_t)ECHUNK * KMEM * NHID * 4);

    k_init<<<(N_NODES + 255) / 256, 256, 0, stream>>>(y, cnt, breg);

    size_t nh = (size_t)N_NODES * N_EDGES;
    k_member<<<(unsigned)((nh + 255) / 256), 256, 0, stream>>>(H, cnt, idx);

    k_convw<<<(NHID * FDIM + 255) / 256, 256, 0, stream>>>(W1, Wres, W1t, Wrt);

    k_gather<<<N_EDGES * KMEM / 4, 256, 0, stream>>>(x, idx, cnt, xg);

    for (int c = 0; c < NCHUNK; ++c) {
        int e0 = c * ECHUNK;
        k_gemm<<<dim3(16, ECHUNK), 256, 0, stream>>>(xg, W1t, Wrt, b1, bres, sf, res, e0);
        k_attn<<<ECHUNK, 256, 0, stream>>>(sf, res, a1, ba1, a2, ba2, Wreg, idx, cnt, y, e0);
    }
}
